// OBBPoseHead_29815662968886
// MI455X (gfx1250) — matrix operand order repaired and checked by reading, not compiled, not run on hardware
//
#include <hip/hip_runtime.h>
#include <math.h>

// CDNA5 / gfx1250: wave32, WMMA 16x16x32 bf16 -> f32 accumulate.
typedef __attribute__((ext_vector_type(16))) __bf16 v16bf;
typedef __attribute__((ext_vector_type(8)))  float  v8f;

__device__ __forceinline__ float fast_silu(float x) {
  // x * sigmoid(x) with hardware v_rcp_f32 (no IEEE div sequence)
  return x * __builtin_amdgcn_rcpf(1.0f + __expf(-x));
}

// ---------------------------------------------------------------------------
// Prep kernels
// ---------------------------------------------------------------------------

// fp32 NCHW -> bf16 NCHW with 1-px halo and right pad so rows stay 16B aligned:
// dst[c][ph][pw], PH = H+2, PW = W+8, interior at (ph-1, pw-1).
template<int CIN, int LOG2W, int LOG2HW>
__global__ __launch_bounds__(256)
void pad_bf16_kernel(const float* __restrict__ src, unsigned short* __restrict__ dst) {
  constexpr int W  = 1 << LOG2W;
  constexpr int H  = 1 << (LOG2HW - LOG2W);
  constexpr int PH = H + 2, PW = W + 8;
  const int total = 8 * CIN * PH * PW;
  int i = blockIdx.x * 256 + threadIdx.x;
  if (i >= total) return;
  int pw = i % PW;
  int t  = i / PW;
  int ph = t % PH;
  int c  = t / PH;                 // c = b*CIN + ci
  int h = ph - 1, w = pw - 1;
  float v = 0.0f;
  if ((unsigned)h < (unsigned)H && (unsigned)w < (unsigned)W)
    v = src[((size_t)c << LOG2HW) + (h << LOG2W) + w];
  ((__bf16*)dst)[i] = (__bf16)v;
}

// w1 (co,ci,3,3) f32  ->  w1p[co][(tap)*CIN + ci] bf16   (K = 9*CIN)
template<int CIN>
__global__ __launch_bounds__(256)
void pack_w1_kernel(const float* __restrict__ w1, unsigned short* __restrict__ w1p) {
  constexpr int K = 9 * CIN;
  int i = blockIdx.x * 256 + threadIdx.x;
  if (i >= CIN * K) return;
  int co = i / K;
  int k  = i - co * K;
  int r  = k / CIN;
  int ci = k & (CIN - 1);
  ((__bf16*)w1p)[i] = (__bf16)w1[(co * CIN + ci) * 9 + r];
}

// w2 (cout,cin) f32 -> w2p[64][CIN] bf16, rows >= cout zero-padded
template<int CIN>
__global__ __launch_bounds__(256)
void pack_w2_kernel(const float* __restrict__ w2, unsigned short* __restrict__ w2p, int cout) {
  int i = blockIdx.x * 256 + threadIdx.x;
  if (i >= 64 * CIN) return;
  int co = i / CIN, ci = i & (CIN - 1);
  ((__bf16*)w2p)[i] = (co < cout) ? (__bf16)w2[co * CIN + ci] : (__bf16)0.0f;
}

// Fold conv bias + eval BN into per-channel scale/shift:
//   y = (conv + b1) * inv + (beta - m*inv)  ==  conv*sc + sh
__global__ __launch_bounds__(256)
void bn_prep_kernel(const float* __restrict__ g, const float* __restrict__ beta,
                    const float* __restrict__ m, const float* __restrict__ v,
                    const float* __restrict__ b1, float* __restrict__ sc,
                    float* __restrict__ sh, int cin) {
  int i = blockIdx.x * 256 + threadIdx.x;
  if (i < cin) {
    float inv = g[i] * rsqrtf(v[i] + 1e-5f);
    sc[i] = inv;
    sh[i] = b1[i] * inv + beta[i] - m[i] * inv;
  }
}

// ---------------------------------------------------------------------------
// Stage 1: Conv3x3 (implicit GEMM on halo-padded bf16 input)
//          + bias/BN/SiLU fused epilogue -> bf16 y (unpadded NCHW)
//   M = CIN, K = 9*CIN, N = B*H*W
//   Block tile: 64(M) x 128(N) x 32(K); 8 waves, each 16(M) x 64(N) = 4 WMMAs
//   Double-buffered LDS; A tile via global_load_async_to_lds_b128 DMA;
//   next-iteration B staged through registers -> one barrier per K-step.
// ---------------------------------------------------------------------------
template<int CIN, int LOG2W, int LOG2HW>
__global__ __launch_bounds__(256)
void conv3x3_bn_silu_kernel(const unsigned short* __restrict__ inp_u,   // padded bf16
                            const unsigned short* __restrict__ w1p_u,   // [CIN][9*CIN] bf16
                            const float* __restrict__ sc,
                            const float* __restrict__ sh,
                            unsigned short* __restrict__ y_u)           // [B][CIN][H][W] bf16
{
  constexpr int LOG2CIN = (CIN == 256) ? 8 : ((CIN == 512) ? 9 : 10);
  constexpr int K    = 9 * CIN;
  constexpr int W    = 1 << LOG2W;
  constexpr int H    = 1 << (LOG2HW - LOG2W);
  constexpr int HW   = 1 << LOG2HW;
  constexpr int PW   = W + 8;
  constexpr int PH   = H + 2;
  constexpr int PHPW = PH * PW;
  constexpr int BM = 64, BN = 128, BK = 32;
  constexpr int NIT = K / BK;

  const __bf16* inp = (const __bf16*)inp_u;
  const __bf16* w1p = (const __bf16*)w1p_u;
  __bf16*       y   = (__bf16*)y_u;

  __shared__ alignas(16) __bf16 As[2][BM * BK];   // [m][k]
  __shared__ alignas(16) __bf16 Bs[2][BN * BK];   // [n][k]

  const int t    = threadIdx.x;
  const int lane = t & 31;
  const int wave = t >> 5;
  const int mw   = (wave & 3) * 16;
  const int nw   = (wave >> 2) * 64;
  const int mtile = blockIdx.y * BM;
  const int ntile = blockIdx.x * BN;

  // A-tile async copy assignment: 16B per thread per buffer
  const int am = t >> 2;
  const int ak = (t & 3) * 8;
  const unsigned lds_a[2] = {
      (unsigned)(uintptr_t)(&As[0][0] + am * BK + ak),
      (unsigned)(uintptr_t)(&As[1][0] + am * BK + ak)};
  const __bf16* arow = w1p + (size_t)(mtile + am) * K + ak;

  // B-tile gather assignment: 16 consecutive n (one row run) for one k-slot
  const int bkl = t >> 3;
  const int bn  = (t & 7) * 16;
  const int n0  = ntile + bn;
  const int b0  = n0 >> LOG2HW;
  const int hh  = (n0 >> LOG2W) & (H - 1);
  const int ww  = n0 & (W - 1);
  const size_t base0 = ((size_t)(b0 * CIN) * PH + (hh + 1)) * PW + (ww + 1);

  auto bsrc = [&](int it) -> const __bf16* {
    const int k0 = it * BK;
    const int r  = k0 >> LOG2CIN;            // wave-uniform filter tap 0..8
    const int dy = r / 3 - 1;
    const int dx = r % 3 - 1;
    const int ci = (k0 & (CIN - 1)) + bkl;
    return inp + base0 + (size_t)ci * PHPW + dy * PW + dx;
  };

  v8f acc[4];
  #pragma unroll
  for (int j = 0; j < 4; ++j)
    #pragma unroll
    for (int e = 0; e < 8; ++e) acc[j][e] = 0.0f;

  // ---- prologue: fill buffer 0 ----
  asm volatile("global_load_async_to_lds_b128 %0, %1, off"
               :: "v"(lds_a[0]), "v"(arow) : "memory");
  {
    const __bf16* src = bsrc(0);
    #pragma unroll
    for (int i = 0; i < 16; ++i) Bs[0][(bn + i) * BK + bkl] = src[i];
  }
  asm volatile("s_wait_asynccnt 0" ::: "memory");
  __syncthreads();

  for (int it = 0; it < NIT; ++it) {
    const int cur = it & 1, nxt = cur ^ 1;
    const bool havenext = (it + 1 < NIT);

    // ---- prefetch next tile: async A DMA + B into registers ----
    __bf16 breg[16];
    if (havenext) {
      asm volatile("global_load_async_to_lds_b128 %0, %1, off"
                   :: "v"(lds_a[nxt]), "v"(arow + (it + 1) * BK) : "memory");
      const __bf16* src = bsrc(it + 1);
      #pragma unroll
      for (int i = 0; i < 16; ++i) breg[i] = src[i];
    }

    // ---- compute on current buffer ----
    v16bf afrag;
    {
      const __bf16* ap = &As[cur][0] + (mw + (lane & 15)) * BK + ((lane >> 4) << 3);
      ((uint4*)&afrag)[0] = *(const uint4*)ap;
      ((uint4*)&afrag)[1] = *(const uint4*)(ap + 16);
    }
    #pragma unroll
    for (int j = 0; j < 4; ++j) {
      v16bf bfrag;
      const __bf16* bp = &Bs[cur][0] + (nw + j * 16 + (lane & 15)) * BK + ((lane >> 4) << 3);
      ((uint4*)&bfrag)[0] = *(const uint4*)bp;
      ((uint4*)&bfrag)[1] = *(const uint4*)(bp + 16);
      acc[j] = __builtin_amdgcn_wmma_f32_16x16x32_bf16(
          false, afrag, false, bfrag, (short)0, acc[j], false, false);
    }

    // ---- commit next B tile, drain async A, single barrier ----
    if (havenext) {
      #pragma unroll
      for (int i = 0; i < 16; ++i) Bs[nxt][(bn + i) * BK + bkl] = breg[i];
      asm volatile("s_wait_asynccnt 0" ::: "memory");
    }
    __syncthreads();
  }

  // ---- fused epilogue: BN scale/shift (bias folded) + fast SiLU -> bf16 ----
  float scv[8], shv[8];
  #pragma unroll
  for (int vv = 0; vv < 8; ++vv) {
    const int m = mtile + mw + vv + ((lane >> 4) << 3);
    scv[vv] = sc[m];
    shv[vv] = sh[m];
  }
  #pragma unroll
  for (int j = 0; j < 4; ++j) {
    const int ncol = ntile + nw + j * 16 + (lane & 15);
    const int b    = ncol >> LOG2HW;
    const int hw   = ncol & (HW - 1);
    #pragma unroll
    for (int vv = 0; vv < 8; ++vv) {
      const int m = mtile + mw + vv + ((lane >> 4) << 3);
      float x = acc[j][vv] * scv[vv] + shv[vv];
      y[(((size_t)(b * CIN + m)) << LOG2HW) + hw] = (__bf16)fast_silu(x);
    }
  }
}

// ---------------------------------------------------------------------------
// Stage 2: Conv1x1 + bias (WMMA GEMM, M=64 padded rows) -> fp32 output slab
// ---------------------------------------------------------------------------
template<int CIN, int LOG2HW, int COUT>
__global__ __launch_bounds__(256)
void conv1x1_bias_kernel(const unsigned short* __restrict__ y_u,     // [B][CIN][HW] bf16
                         const unsigned short* __restrict__ w2p_u,   // [64][CIN]    bf16
                         const float* __restrict__ b2,
                         float* __restrict__ out)                    // [B][COUT][HW] fp32
{
  constexpr int HW = 1 << LOG2HW;
  constexpr int BN = 128, BK = 32;
  constexpr int NIT = CIN / BK;

  const __bf16* yb  = (const __bf16*)y_u;
  const __bf16* w2p = (const __bf16*)w2p_u;

  __shared__ alignas(16) __bf16 As[2][64 * BK];
  __shared__ alignas(16) __bf16 Bs[2][BN * BK];

  const int t    = threadIdx.x;
  const int lane = t & 31;
  const int wave = t >> 5;
  const int mw   = (wave & 3) * 16;
  const int nw   = (wave >> 2) * 64;
  const int ntile = blockIdx.x * BN;

  const int am  = t >> 2;
  const int ak  = (t & 3) * 8;
  const unsigned lds_a[2] = {
      (unsigned)(uintptr_t)(&As[0][0] + am * BK + ak),
      (unsigned)(uintptr_t)(&As[1][0] + am * BK + ak)};
  const __bf16* arow = w2p + (size_t)am * CIN + ak;

  const int bkl = t >> 3;
  const int bn  = (t & 7) * 16;
  const int n0  = ntile + bn;
  const int b0  = n0 >> LOG2HW;
  const int hw0 = n0 & (HW - 1);               // 16-run never crosses an image
  const __bf16* brow = yb + (((size_t)(b0 * CIN)) << LOG2HW) + hw0;

  v8f acc[4];
  #pragma unroll
  for (int j = 0; j < 4; ++j)
    #pragma unroll
    for (int e = 0; e < 8; ++e) acc[j][e] = 0.0f;

  union BV { uint4 u[2]; __bf16 e[16]; };

  // ---- prologue: fill buffer 0 ----
  asm volatile("global_load_async_to_lds_b128 %0, %1, off"
               :: "v"(lds_a[0]), "v"(arow) : "memory");
  {
    BV tmp;
    const __bf16* src = brow + (((size_t)bkl) << LOG2HW);
    tmp.u[0] = ((const uint4*)src)[0];
    tmp.u[1] = ((const uint4*)src)[1];
    #pragma unroll
    for (int i = 0; i < 16; ++i) Bs[0][(bn + i) * BK + bkl] = tmp.e[i];
  }
  asm volatile("s_wait_asynccnt 0" ::: "memory");
  __syncthreads();

  for (int it = 0; it < NIT; ++it) {
    const int cur = it & 1, nxt = cur ^ 1;
    const bool havenext = (it + 1 < NIT);

    BV tmp;
    if (havenext) {
      asm volatile("global_load_async_to_lds_b128 %0, %1, off"
                   :: "v"(lds_a[nxt]), "v"(arow + (it + 1) * BK) : "memory");
      const __bf16* src = brow + (((size_t)((it + 1) * BK + bkl)) << LOG2HW);
      tmp.u[0] = ((const uint4*)src)[0];
      tmp.u[1] = ((const uint4*)src)[1];
    }

    v16bf afrag;
    {
      const __bf16* ap = &As[cur][0] + (mw + (lane & 15)) * BK + ((lane >> 4) << 3);
      ((uint4*)&afrag)[0] = *(const uint4*)ap;
      ((uint4*)&afrag)[1] = *(const uint4*)(ap + 16);
    }
    #pragma unroll
    for (int j = 0; j < 4; ++j) {
      v16bf bfrag;
      const __bf16* bp = &Bs[cur][0] + (nw + j * 16 + (lane & 15)) * BK + ((lane >> 4) << 3);
      ((uint4*)&bfrag)[0] = *(const uint4*)bp;
      ((uint4*)&bfrag)[1] = *(const uint4*)(bp + 16);
      acc[j] = __builtin_amdgcn_wmma_f32_16x16x32_bf16(
          false, afrag, false, bfrag, (short)0, acc[j], false, false);
    }

    if (havenext) {
      #pragma unroll
      for (int i = 0; i < 16; ++i) Bs[nxt][(bn + i) * BK + bkl] = tmp.e[i];
      asm volatile("s_wait_asynccnt 0" ::: "memory");
    }
    __syncthreads();
  }

  #pragma unroll
  for (int j = 0; j < 4; ++j) {
    const int ncol = ntile + nw + j * 16 + (lane & 15);
    const int b    = ncol >> LOG2HW;
    const int hw   = ncol & (HW - 1);
    #pragma unroll
    for (int vv = 0; vv < 8; ++vv) {
      const int m = mw + vv + ((lane >> 4) << 3);
      if (m < COUT)
        out[(((size_t)(b * COUT + m)) << LOG2HW) + hw] = acc[j][vv] + b2[m];
    }
  }
}

// ---------------------------------------------------------------------------
// Host launcher
// ---------------------------------------------------------------------------
extern "C" void kernel_launch(void* const* d_in, const int* in_sizes, int n_in,
                              void* d_out, int out_size, void* d_ws, size_t ws_size,
                              hipStream_t stream) {
  (void)in_sizes; (void)n_in; (void)out_size; (void)ws_size;

  static const int    kCIN[3]    = {256, 512, 1024};
  static const int    kHW[3]     = {16384, 4096, 1024};
  static const int    kPH[3]     = {130, 66, 34};    // H+2
  static const int    kPW[3]     = {136, 72, 40};    // W+8
  // heads: det3, det4, det5, kp3, kp4, kp5
  static const int    kLvl[6]    = {0, 1, 2, 0, 1, 2};
  static const int    kCOUT[6]   = {53, 53, 53, 3, 3, 3};
  static const size_t kOutOff[6] = {0, 6946816, 8683520, 9117696, 9510912, 9609216};

  const float* pin[3] = {(const float*)d_in[0], (const float*)d_in[1], (const float*)d_in[2]};
  // JAX pytree (sorted dict) leaf order per head: b1,b2,beta,g,m,v,w1,w2
  auto P = [&](int h, int j) -> const float* { return (const float*)d_in[3 + h * 8 + j]; };

  // ---- workspace layout ----
  char* ws = (char*)d_ws;
  size_t off = 0;
  auto take = [&](size_t bytes) { size_t o = off; off += (bytes + 255) & ~(size_t)255; return o; };

  size_t inp_off[3];
  for (int l = 0; l < 3; ++l)
    inp_off[l] = take((size_t)8 * kCIN[l] * kPH[l] * kPW[l] * 2);
  size_t w1p_off[6], w2p_off[6], sc_off[6], sh_off[6];
  for (int h = 0; h < 6; ++h) {
    const int c = kCIN[kLvl[h]];
    w1p_off[h] = take((size_t)c * 9 * c * 2);
    w2p_off[h] = take((size_t)64 * c * 2);
    sc_off[h]  = take((size_t)c * 4);
    sh_off[h]  = take((size_t)c * 4);
  }
  size_t y_off = take((size_t)8 * 256 * 16384 * 2);   // reused across heads (stream-ordered)

  // ---- 1) activations fp32 -> halo-padded bf16 (L2-resident operand) ----
  {
    int n0 = 8 * kCIN[0] * kPH[0] * kPW[0];
    pad_bf16_kernel<256, 7, 14><<<(n0 + 255) / 256, 256, 0, stream>>>(
        pin[0], (unsigned short*)(ws + inp_off[0]));
    int n1 = 8 * kCIN[1] * kPH[1] * kPW[1];
    pad_bf16_kernel<512, 6, 12><<<(n1 + 255) / 256, 256, 0, stream>>>(
        pin[1], (unsigned short*)(ws + inp_off[1]));
    int n2 = 8 * kCIN[2] * kPH[2] * kPW[2];
    pad_bf16_kernel<1024, 5, 10><<<(n2 + 255) / 256, 256, 0, stream>>>(
        pin[2], (unsigned short*)(ws + inp_off[2]));
  }

  // ---- 2) per-head weight repack + BN constant folding ----
  for (int h = 0; h < 6; ++h) {
    const int lv = kLvl[h], c = kCIN[lv];
    bn_prep_kernel<<<(c + 255) / 256, 256, 0, stream>>>(
        P(h, 3), P(h, 2), P(h, 4), P(h, 5), P(h, 0),
        (float*)(ws + sc_off[h]), (float*)(ws + sh_off[h]), c);
    const int n1 = c * 9 * c, n2 = 64 * c;
    unsigned short* w1p = (unsigned short*)(ws + w1p_off[h]);
    unsigned short* w2p = (unsigned short*)(ws + w2p_off[h]);
    if (lv == 0) {
      pack_w1_kernel<256 ><<<(n1 + 255) / 256, 256, 0, stream>>>(P(h, 6), w1p);
      pack_w2_kernel<256 ><<<(n2 + 255) / 256, 256, 0, stream>>>(P(h, 7), w2p, kCOUT[h]);
    } else if (lv == 1) {
      pack_w1_kernel<512 ><<<(n1 + 255) / 256, 256, 0, stream>>>(P(h, 6), w1p);
      pack_w2_kernel<512 ><<<(n2 + 255) / 256, 256, 0, stream>>>(P(h, 7), w2p, kCOUT[h]);
    } else {
      pack_w1_kernel<1024><<<(n1 + 255) / 256, 256, 0, stream>>>(P(h, 6), w1p);
      pack_w2_kernel<1024><<<(n2 + 255) / 256, 256, 0, stream>>>(P(h, 7), w2p, kCOUT[h]);
    }
  }

  // ---- 3) per head: fused conv3x3+BN+SiLU (WMMA) then conv1x1+bias (WMMA) ----
  for (int h = 0; h < 6; ++h) {
    const int lv = kLvl[h];
    const int ntot = 8 * kHW[lv];
    const unsigned short* inp = (const unsigned short*)(ws + inp_off[lv]);
    const unsigned short* w1p = (const unsigned short*)(ws + w1p_off[h]);
    const unsigned short* w2p = (const unsigned short*)(ws + w2p_off[h]);
    const float* sc = (const float*)(ws + sc_off[h]);
    const float* sh = (const float*)(ws + sh_off[h]);
    unsigned short* ybuf = (unsigned short*)(ws + y_off);
    float* outp = (float*)d_out + kOutOff[h];
    const float* b2 = P(h, 1);

    dim3 blk(256);
    dim3 g1(ntot / 128, kCIN[lv] / 64);
    dim3 g2(ntot / 128, 1);

    if (lv == 0) {
      conv3x3_bn_silu_kernel<256, 7, 14><<<g1, blk, 0, stream>>>(inp, w1p, sc, sh, ybuf);
      if (kCOUT[h] == 53)
        conv1x1_bias_kernel<256, 14, 53><<<g2, blk, 0, stream>>>(ybuf, w2p, b2, outp);
      else
        conv1x1_bias_kernel<256, 14, 3 ><<<g2, blk, 0, stream>>>(ybuf, w2p, b2, outp);
    } else if (lv == 1) {
      conv3x3_bn_silu_kernel<512, 6, 12><<<g1, blk, 0, stream>>>(inp, w1p, sc, sh, ybuf);
      if (kCOUT[h] == 53)
        conv1x1_bias_kernel<512, 12, 53><<<g2, blk, 0, stream>>>(ybuf, w2p, b2, outp);
      else
        conv1x1_bias_kernel<512, 12, 3 ><<<g2, blk, 0, stream>>>(ybuf, w2p, b2, outp);
    } else {
      conv3x3_bn_silu_kernel<1024, 5, 10><<<g1, blk, 0, stream>>>(inp, w1p, sc, sh, ybuf);
      if (kCOUT[h] == 53)
        conv1x1_bias_kernel<1024, 10, 53><<<g2, blk, 0, stream>>>(ybuf, w2p, b2, outp);
      else
        conv1x1_bias_kernel<1024, 10, 3 ><<<g2, blk, 0, stream>>>(ybuf, w2p, b2, outp);
    }
  }
}
